// Attention_62749472195138
// MI455X (gfx1250) — matrix operand order repaired and checked by reading, not compiled, not run on hardware
//
#include <hip/hip_runtime.h>
#include <hip/hip_bf16.h>
#include <math.h>

typedef __attribute__((ext_vector_type(16))) _Float16 v16h;
typedef __attribute__((ext_vector_type(8)))  _Float16 v8h;
typedef __attribute__((ext_vector_type(8)))  float    v8f;

#define WMMA_F16(a, b, c) \
  __builtin_amdgcn_wmma_f32_16x16x32_f16(false, (a), false, (b), (short)0, (c), false, false)

// CDNA5 async global->LDS copy (ASYNCcnt-tracked, per-lane 16B)
__device__ __forceinline__ void async_ld_b128(uint32_t lds_byte_off, const void* gptr) {
  asm volatile("global_load_async_to_lds_b128 %0, %1, off"
               :: "v"(lds_byte_off), "v"(gptr)
               : "memory");
}
__device__ __forceinline__ uint32_t lds_off(const void* p) {
  // generic pointers to LDS carry the LDS byte offset in addr[31:0]
  return (uint32_t)(uintptr_t)p;
}
#define S_WAIT_ASYNC(N) asm volatile("s_wait_asynccnt " #N ::: "memory")

static constexpr int BATCH = 8;
static constexpr int NTOK  = 1024;
static constexpr int CH    = 512;
static constexpr int HEADS = 8;
static constexpr int DH    = 64;
static constexpr int F3    = 1536;

// ---------------------------------------------------------------------------
// K0: convert weights to f16, transposed (output-feature-major, K contiguous)
// ---------------------------------------------------------------------------
__global__ void convert_weights(const float* __restrict__ wqkv,
                                const float* __restrict__ wout,
                                _Float16* __restrict__ wqkvT,
                                _Float16* __restrict__ woutT) {
  int i = blockIdx.x * blockDim.x + threadIdx.x;
  const int n1 = F3 * CH;
  const int n2 = CH * CH;
  if (i < n1) {
    int f = i >> 9, c = i & 511;
    wqkvT[i] = (_Float16)wqkv[c * F3 + f];
  } else if (i < n1 + n2) {
    int j = i - n1;
    int co = j >> 9, f = j & 511;
    woutT[j] = (_Float16)wout[f * CH + co];
  }
}

// ---------------------------------------------------------------------------
// K1: LayerNorm over channels (scale only), emit f16
// ---------------------------------------------------------------------------
__global__ __launch_bounds__(256) void layernorm_f16(const float* __restrict__ x,
                                                     const float* __restrict__ g,
                                                     _Float16* __restrict__ y) {
  __shared__ float s1[256], s2[256];
  const int row = blockIdx.x, tid = threadIdx.x;
  const float* xr = x + (size_t)row * CH;
  float a0 = xr[tid], a1 = xr[tid + 256];
  s1[tid] = a0 + a1;
  s2[tid] = a0 * a0 + a1 * a1;
  __syncthreads();
  for (int off = 128; off > 0; off >>= 1) {
    if (tid < off) { s1[tid] += s1[tid + off]; s2[tid] += s2[tid + off]; }
    __syncthreads();
  }
  float mean = s1[0] * (1.0f / CH);
  float var  = s2[0] * (1.0f / CH) - mean * mean;
  float rstd = rsqrtf(var + 1e-5f);
  _Float16* yr = y + (size_t)row * CH;
  yr[tid]       = (_Float16)((a0 - mean) * rstd * g[tid]);
  yr[tid + 256] = (_Float16)((a1 - mean) * rstd * g[tid + 256]);
}

// ---------------------------------------------------------------------------
// K2: QKV GEMM with async-LDS-staged B. WG tile 128 rows x 128 cols.
// 8 waves x (16 rows x 128 cols). B k-slab (128x32 f16 = 8KB) double-buffered.
// ---------------------------------------------------------------------------
__global__ __launch_bounds__(256) void gemm_qkv(const _Float16* __restrict__ A,
                                                const _Float16* __restrict__ BT,
                                                _Float16* __restrict__ qh,
                                                _Float16* __restrict__ kh,
                                                _Float16* __restrict__ vt) {
  __shared__ _Float16 bstage[2][128][32];   // 16 KB
  const int wave = threadIdx.x >> 5;
  const int m    = threadIdx.x & 15;
  const int half = (threadIdx.x >> 4) & 1;
  const int colBlk = blockIdx.x * 128;
  const int rowBlk = blockIdx.y * 128 + wave * 16;

  auto stage = [&](int kc, int pb) {
#pragma unroll
    for (int t = 0; t < 2; ++t) {
      int c = threadIdx.x + t * 256;      // 512 chunks of 16B
      int col = c >> 2, seg = c & 3;
      async_ld_b128(lds_off(&bstage[pb][col][seg * 8]),
                    BT + (size_t)(colBlk + col) * CH + kc + seg * 8);
    }
  };

  const v8f vz = {0.f, 0.f, 0.f, 0.f, 0.f, 0.f, 0.f, 0.f};
  v8f acc[8] = {vz, vz, vz, vz, vz, vz, vz, vz};
  const _Float16* arow = A + (size_t)(rowBlk + m) * CH;

  stage(0, 0);
  for (int it = 0; it < 16; ++it) {
    const int kc = it * 32, pb = it & 1;
    if (it + 1 < 16) { stage(kc + 32, pb ^ 1); S_WAIT_ASYNC(2); }
    else             { S_WAIT_ASYNC(0); }
    __syncthreads();
    union { v16h v; v8h h2[2]; } ua;
    ua.h2[0] = *(const v8h*)(arow + kc + 16 * half);
    ua.h2[1] = *(const v8h*)(arow + kc + 16 * half + 8);
#pragma unroll
    for (int g2 = 0; g2 < 8; ++g2) {
      v16h bv = *(const v16h*)(&bstage[pb][g2 * 16 + m][16 * half]);
      acc[g2] = WMMA_F16(ua.v, bv, acc[g2]);
    }
    __syncthreads();
  }

#pragma unroll
  for (int g2 = 0; g2 < 8; ++g2) {
#pragma unroll
    for (int r = 0; r < 8; ++r) {
      int rowo = rowBlk + 8 * half + r;
      int f    = colBlk + g2 * 16 + m;
      float val = acc[g2][r];
      int b = rowo >> 10, t = rowo & 1023;
      if (f < 512) {
        int h = f >> 6, d = f & 63;
        qh[(((size_t)(b * HEADS + h) * NTOK + t) << 6) + d] = (_Float16)val;
      } else if (f < 1024) {
        int f2 = f - 512, h = f2 >> 6, d = f2 & 63;
        kh[(((size_t)(b * HEADS + h) * NTOK + t) << 6) + d] = (_Float16)val;
      } else {
        int f2 = f - 1024, h = f2 >> 6, d = f2 & 63;
        vt[(((size_t)(b * HEADS + h) * DH + d) << 10) + t] = (_Float16)val;
      }
    }
  }
}

// ---------------------------------------------------------------------------
// K3: token-axis l2 norms (axis=1 quirk). scale_q folds SCALE=10.
// ---------------------------------------------------------------------------
__global__ __launch_bounds__(256) void col_norms(const _Float16* __restrict__ qh,
                                                 const _Float16* __restrict__ kh,
                                                 float* __restrict__ sq,
                                                 float* __restrict__ sk) {
  __shared__ float red[256];
  const int col = blockIdx.x;
  const int b   = blockIdx.y;
  const int which = blockIdx.z;
  const _Float16* src = which ? kh : qh;
  const int h = col >> 6, d = col & 63;
  const _Float16* base = src + (((size_t)(b * HEADS + h) * NTOK) << 6) + d;
  float sum = 0.f;
  for (int t = threadIdx.x; t < NTOK; t += 256) {
    float v = (float)base[(size_t)t * DH];
    sum += v * v;
  }
  red[threadIdx.x] = sum;
  __syncthreads();
  for (int off = 128; off > 0; off >>= 1) {
    if (threadIdx.x < off) red[threadIdx.x] += red[threadIdx.x + off];
    __syncthreads();
  }
  if (threadIdx.x == 0) {
    float nrm = sqrtf(red[0]);
    float scl = 1.0f / fmaxf(nrm, 1e-12f);
    if (which) sk[b * CH + col] = scl;
    else       sq[b * CH + col] = 10.0f * scl;
  }
}

// ---------------------------------------------------------------------------
// K4: apply per-(b,h,d) column scales to q and k in place
// ---------------------------------------------------------------------------
__global__ void apply_scales(_Float16* __restrict__ qh, _Float16* __restrict__ kh,
                             const float* __restrict__ sq, const float* __restrict__ sk) {
  size_t i = (size_t)blockIdx.x * blockDim.x + threadIdx.x;
  int d = (int)(i & 63);
  int h = (int)((i >> 16) & 7);
  int b = (int)(i >> 19);
  int col = h * 64 + d;
  qh[i] = (_Float16)((float)qh[i] * sq[b * CH + col]);
  kh[i] = (_Float16)((float)kh[i] * sk[b * CH + col]);
}

// ---------------------------------------------------------------------------
// K5: flash attention per (b,h). K/V chunks async-staged to LDS and shared
// by all 8 waves (double-buffered). Each wave: 16x64 Q tile.
// ---------------------------------------------------------------------------
__global__ __launch_bounds__(256) void flash_attn(const _Float16* __restrict__ qh,
                                                  const _Float16* __restrict__ kh,
                                                  const _Float16* __restrict__ vt,
                                                  _Float16* __restrict__ ao) {
  __shared__ _Float16 kbuf[2][32][64];   // 8 KB  (token-rel x d)
  __shared__ _Float16 vbuf[2][64][32];   // 8 KB  (d x token-rel, from vT)
  __shared__ _Float16 pbuf[8][16][32];   // 8 KB  per-wave P tiles
  const int wave = threadIdx.x >> 5;
  const int m    = threadIdx.x & 15;
  const int half = (threadIdx.x >> 4) & 1;
  const int bh = blockIdx.x;
  const int b  = bh >> 3, h = bh & 7;
  const int row0 = blockIdx.y * 128 + wave * 16;

  const _Float16* qp = qh + (size_t)bh * NTOK * DH;
  const _Float16* kp = kh + (size_t)bh * NTOK * DH;
  const _Float16* vp = vt + (size_t)bh * DH * NTOK;

  auto stage = [&](int j0, int pb) {
    {
      int row = threadIdx.x >> 3, seg = threadIdx.x & 7;     // 32 x 8 chunks
      async_ld_b128(lds_off(&kbuf[pb][row][seg * 8]),
                    kp + (size_t)(j0 + row) * DH + seg * 8);
    }
    {
      int row = threadIdx.x >> 2, seg = threadIdx.x & 3;     // 64 x 4 chunks
      async_ld_b128(lds_off(&vbuf[pb][row][seg * 8]),
                    vp + (size_t)row * NTOK + j0 + seg * 8);
    }
  };

  union { v16h v; v8h h2[2]; } aq0, aq1, ap;
  const _Float16* qr = qp + (size_t)(row0 + m) * DH;
  aq0.h2[0] = *(const v8h*)(qr + 16 * half);
  aq0.h2[1] = *(const v8h*)(qr + 8 + 16 * half);
  aq1.h2[0] = *(const v8h*)(qr + 32 + 16 * half);
  aq1.h2[1] = *(const v8h*)(qr + 40 + 16 * half);

  const v8f vz = {0.f, 0.f, 0.f, 0.f, 0.f, 0.f, 0.f, 0.f};
  v8f O[4] = {vz, vz, vz, vz};
  float M[8], L[8];
#pragma unroll
  for (int r = 0; r < 8; ++r) { M[r] = -1e30f; L[r] = 0.f; }

  stage(0, 0);
  for (int c = 0; c < NTOK / 32; ++c) {
    const int j0 = c * 32, pb = c & 1;
    if (c + 1 < NTOK / 32) { stage(j0 + 32, pb ^ 1); S_WAIT_ASYNC(2); }
    else                   { S_WAIT_ASYNC(0); }
    __syncthreads();

    // ---- S = Q (16x64) * K^T (64x32) from LDS ----
    v8f s0 = vz, s1 = vz;
    {
      v16h b00 = *(const v16h*)(&kbuf[pb][m][16 * half]);
      v16h b01 = *(const v16h*)(&kbuf[pb][m][32 + 16 * half]);
      v16h b10 = *(const v16h*)(&kbuf[pb][16 + m][16 * half]);
      v16h b11 = *(const v16h*)(&kbuf[pb][16 + m][32 + 16 * half]);
      s0 = WMMA_F16(aq0.v, b00, s0);
      s0 = WMMA_F16(aq1.v, b01, s0);
      s1 = WMMA_F16(aq0.v, b10, s1);
      s1 = WMMA_F16(aq1.v, b11, s1);
    }
    // ---- online softmax; row (8*half + r), cols across 16-lane half ----
#pragma unroll
    for (int r = 0; r < 8; ++r) {
      float pm = fmaxf(s0[r], s1[r]);
#pragma unroll
      for (int off = 1; off < 16; off <<= 1) pm = fmaxf(pm, __shfl_xor(pm, off, 32));
      float mn = fmaxf(M[r], pm);
      float cc = __expf(M[r] - mn);
      M[r] = mn;
      float p0 = __expf(s0[r] - mn);
      float p1 = __expf(s1[r] - mn);
      pbuf[wave][8 * half + r][m]      = (_Float16)p0;
      pbuf[wave][8 * half + r][16 + m] = (_Float16)p1;
      float sump = p0 + p1;
#pragma unroll
      for (int off = 1; off < 16; off <<= 1) sump += __shfl_xor(sump, off, 32);
      L[r] = L[r] * cc + sump;
#pragma unroll
      for (int g = 0; g < 4; ++g) O[g][r] *= cc;
    }
    // C-layout -> A-layout via LDS (wave-private; DS in-order per wave)
    asm volatile("s_wait_dscnt 0" ::: "memory");
    const _Float16* pr = &pbuf[wave][m][0];
    ap.h2[0] = *(const v8h*)(pr + 16 * half);
    ap.h2[1] = *(const v8h*)(pr + 8 + 16 * half);
    // ---- O += P (16x32) * V (32x64) from LDS ----
#pragma unroll
    for (int g = 0; g < 4; ++g) {
      v16h bv = *(const v16h*)(&vbuf[pb][g * 16 + m][16 * half]);
      O[g] = WMMA_F16(ap.v, bv, O[g]);
    }
    __syncthreads();   // everyone done with buf pb before it is restaged
  }

  float inv[8];
#pragma unroll
  for (int r = 0; r < 8; ++r) inv[r] = 1.0f / L[r];
#pragma unroll
  for (int g = 0; g < 4; ++g) {
#pragma unroll
    for (int r = 0; r < 8; ++r) {
      int trow = row0 + 8 * half + r;
      size_t idx = (size_t)(b * NTOK + trow) * CH + h * DH + g * 16 + m;
      ao[idx] = (_Float16)(O[g][r] * inv[r]);
    }
  }
}

// ---------------------------------------------------------------------------
// K6: output projection with async-LDS-staged B. WG tile 128 x 128.
// ---------------------------------------------------------------------------
__global__ __launch_bounds__(256) void gemm_out(const _Float16* __restrict__ A,
                                                const _Float16* __restrict__ BT,
                                                const float* __restrict__ bias,
                                                float* __restrict__ out) {
  __shared__ _Float16 bstage[2][128][32];   // 16 KB
  const int wave = threadIdx.x >> 5;
  const int m    = threadIdx.x & 15;
  const int half = (threadIdx.x >> 4) & 1;
  const int colBlk = blockIdx.x * 128;
  const int rowBlk = blockIdx.y * 128 + wave * 16;

  auto stage = [&](int kc, int pb) {
#pragma unroll
    for (int t = 0; t < 2; ++t) {
      int c = threadIdx.x + t * 256;
      int col = c >> 2, seg = c & 3;
      async_ld_b128(lds_off(&bstage[pb][col][seg * 8]),
                    BT + (size_t)(colBlk + col) * CH + kc + seg * 8);
    }
  };

  const v8f vz = {0.f, 0.f, 0.f, 0.f, 0.f, 0.f, 0.f, 0.f};
  v8f acc[8] = {vz, vz, vz, vz, vz, vz, vz, vz};
  const _Float16* arow = A + (size_t)(rowBlk + m) * CH;

  stage(0, 0);
  for (int it = 0; it < 16; ++it) {
    const int kc = it * 32, pb = it & 1;
    if (it + 1 < 16) { stage(kc + 32, pb ^ 1); S_WAIT_ASYNC(2); }
    else             { S_WAIT_ASYNC(0); }
    __syncthreads();
    union { v16h v; v8h h2[2]; } ua;
    ua.h2[0] = *(const v8h*)(arow + kc + 16 * half);
    ua.h2[1] = *(const v8h*)(arow + kc + 16 * half + 8);
#pragma unroll
    for (int g2 = 0; g2 < 8; ++g2) {
      v16h bv = *(const v16h*)(&bstage[pb][g2 * 16 + m][16 * half]);
      acc[g2] = WMMA_F16(ua.v, bv, acc[g2]);
    }
    __syncthreads();
  }

#pragma unroll
  for (int g2 = 0; g2 < 8; ++g2) {
#pragma unroll
    for (int r = 0; r < 8; ++r) {
      int rowo = rowBlk + 8 * half + r;
      int col  = colBlk + g2 * 16 + m;
      out[(size_t)rowo * CH + col] = acc[g2][r] + bias[col];
    }
  }
}

// ---------------------------------------------------------------------------
extern "C" void kernel_launch(void* const* d_in, const int* in_sizes, int n_in,
                              void* d_out, int out_size, void* d_ws, size_t ws_size,
                              hipStream_t stream) {
  (void)in_sizes; (void)n_in; (void)out_size; (void)ws_size;
  const float* x        = (const float*)d_in[0];
  const float* ln_scale = (const float*)d_in[1];
  const float* w_qkv    = (const float*)d_in[2];
  const float* w_out    = (const float*)d_in[3];
  const float* b_out    = (const float*)d_in[4];
  float* out = (float*)d_out;

  char* ws = (char*)d_ws;
  size_t off = 0;
  auto alloc = [&](size_t bytes) -> char* {
    char* p = ws + off;
    off = (off + bytes + 255) & ~(size_t)255;
    return p;
  };
  _Float16* y_h   = (_Float16*)alloc((size_t)BATCH * NTOK * CH * 2);
  _Float16* wqkvT = (_Float16*)alloc((size_t)F3 * CH * 2);
  _Float16* woutT = (_Float16*)alloc((size_t)CH * CH * 2);
  _Float16* qhb   = (_Float16*)alloc((size_t)BATCH * HEADS * NTOK * DH * 2);
  _Float16* khb   = (_Float16*)alloc((size_t)BATCH * HEADS * NTOK * DH * 2);
  _Float16* vtb   = (_Float16*)alloc((size_t)BATCH * HEADS * DH * NTOK * 2);
  float*    sq    = (float*)alloc((size_t)BATCH * CH * 4);
  float*    sk    = (float*)alloc((size_t)BATCH * CH * 4);
  _Float16* ao    = (_Float16*)alloc((size_t)BATCH * NTOK * CH * 2);

  convert_weights<<<(F3 * CH + CH * CH + 255) / 256, 256, 0, stream>>>(w_qkv, w_out, wqkvT, woutT);
  layernorm_f16<<<BATCH * NTOK, 256, 0, stream>>>(x, ln_scale, y_h);
  gemm_qkv<<<dim3(F3 / 128, BATCH * NTOK / 128), 256, 0, stream>>>(y_h, wqkvT, qhb, khb, vtb);
  col_norms<<<dim3(CH, BATCH, 2), 256, 0, stream>>>(qhb, khb, sq, sk);
  apply_scales<<<(BATCH * HEADS * NTOK * DH) / 256, 256, 0, stream>>>(qhb, khb, sq, sk);
  flash_attn<<<dim3(BATCH * HEADS, NTOK / 128), 256, 0, stream>>>(qhb, khb, vtb, ao);
  gemm_out<<<dim3(CH / 128, BATCH * NTOK / 128), 256, 0, stream>>>(ao, woutT, b_out, out);
}
